// ODEBlock_32512902431439
// MI455X (gfx1250) — matrix operand order repaired and checked by reading, not compiled, not run on hardware
//
#include <hip/hip_runtime.h>
#include <math.h>

typedef __attribute__((ext_vector_type(16))) __bf16 v16bf;
typedef __attribute__((ext_vector_type(8)))  __bf16 v8bf;
typedef __attribute__((ext_vector_type(4)))  __bf16 v4bf;
typedef __attribute__((ext_vector_type(8)))  float  v8f;

#define BM 128
#define BN 128
#define BK 32
#define LDSS 40   // bf16 elements per LDS row (32 data + pad) -> 80B, keeps B128 alignment

__device__ __forceinline__ v16bf cat8(v8bf lo, v8bf hi) {
  return __builtin_shufflevector(lo, hi, 0, 1, 2, 3, 4, 5, 6, 7,
                                 8, 9, 10, 11, 12, 13, 14, 15);
}

// Async 16-byte global -> LDS copy (ASYNCcnt-tracked, no VGPR round-trip).
// Low 32 bits of a generic pointer to __shared__ are the wave-relative LDS
// byte offset (flat-aperture rule), which is what VDST wants.
__device__ __forceinline__ void async_copy16(unsigned lds_byte_off, const void* g) {
  unsigned long long ga = (unsigned long long)g;
  asm volatile("global_load_async_to_lds_b128 %0, %1, off"
               :: "v"(lds_byte_off), "v"(ga)
               : "memory");
}

// Issue one 128x32 A tile + 128x32 Bt tile stage: 4 async B128s per thread.
__device__ __forceinline__ void issue_stage(
    const __bf16* __restrict__ A, const __bf16* __restrict__ Bt, int K,
    int m0, int n0, int kc, unsigned ldsA, unsigned ldsB, int tid) {
#pragma unroll
  for (int it = 0; it < 2; ++it) {
    const int c = tid + it * 256;
    const int row = c >> 2, cc = c & 3;
    const unsigned loff = (unsigned)(row * LDSS + cc * 8) * 2u;
    async_copy16(ldsA + loff, (const void*)&A[(size_t)(m0 + row) * K + kc + cc * 8]);
    async_copy16(ldsB + loff, (const void*)&Bt[(size_t)(n0 + row) * K + kc + cc * 8]);
  }
}

// One K=32 chunk of WMMA work on a staged tile pair.
__device__ __forceinline__ void compute_tile(
    const __bf16* lsA, const __bf16* lsB,
    int wm, int wn, int half, int mr, v8f acc[2][4]) {
  // A fragments (16x32 bf16): lane<16 holds K 0..7 & 16..23, lane>=16 shifted
  v16bf af[2];
#pragma unroll
  for (int i = 0; i < 2; ++i) {
    const __bf16* p = &lsA[(wm * 32 + i * 16 + mr) * LDSS];
    af[i] = cat8(*(const v8bf*)&p[half * 8], *(const v8bf*)&p[16 + half * 8]);
  }
  // B fragments (32x16 bf16): lane<16 holds K 0..7 & 16..23, lane>=16 shifted
  v16bf bfm[4];
#pragma unroll
  for (int j = 0; j < 4; ++j) {
    const __bf16* p = &lsB[(wn * 64 + j * 16 + mr) * LDSS];
    bfm[j] = cat8(*(const v8bf*)&p[half * 8], *(const v8bf*)&p[16 + half * 8]);
  }
#pragma unroll
  for (int i = 0; i < 2; ++i)
#pragma unroll
    for (int j = 0; j < 4; ++j)
      acc[i][j] = __builtin_amdgcn_wmma_f32_16x16x32_bf16(
          false, af[i], false, bfm[j], (short)0, acc[i][j], false, false);
}

// ---------------------------------------------------------------------------
// Double-buffered async-pipelined main loop:
// C(128x128) += A(128xK) * Bt(128xK)^T, bf16 in, f32 acc.
// A row-major [M][K], Bt row-major [N][K] (pre-transposed weights).
// ---------------------------------------------------------------------------
__device__ __forceinline__ void gemm_mainloop(
    const __bf16* __restrict__ A, const __bf16* __restrict__ Bt, int K,
    int m0, int n0, __bf16* ls, v8f acc[2][4]) {
  __bf16* lsA0 = ls;
  __bf16* lsB0 = ls + BM * LDSS;
  __bf16* lsA1 = ls + 2 * BM * LDSS;
  __bf16* lsB1 = ls + 3 * BM * LDSS;
  const unsigned a0 = (unsigned)(unsigned long long)lsA0;
  const unsigned b0 = (unsigned)(unsigned long long)lsB0;
  const unsigned a1 = (unsigned)(unsigned long long)lsA1;
  const unsigned b1 = (unsigned)(unsigned long long)lsB1;

  const int tid = threadIdx.x, lane = tid & 31, wave = tid >> 5;
  const int wm = wave & 3;   // M sub-tile: wm*32
  const int wn = wave >> 2;  // N sub-tile: wn*64
  const int half = lane >> 4, mr = lane & 15;

  issue_stage(A, Bt, K, m0, n0, 0, a0, b0, tid);

  for (int kc = 0; kc < K; kc += 2 * BK) {
    // ---- compute buf0 (chunk kc); issue chunk kc+BK into buf1 ----
    if (kc + BK < K) {
      issue_stage(A, Bt, K, m0, n0, kc + BK, a1, b1, tid);
      asm volatile("s_wait_asynccnt 0x4" ::: "memory");  // buf0 done, buf1 in flight
    } else {
      asm volatile("s_wait_asynccnt 0x0" ::: "memory");
    }
    __syncthreads();
    compute_tile(lsA0, lsB0, wm, wn, half, mr, acc);
    __syncthreads();
    if (kc + BK >= K) break;

    // ---- compute buf1 (chunk kc+BK); issue chunk kc+2*BK into buf0 ----
    if (kc + 2 * BK < K) {
      issue_stage(A, Bt, K, m0, n0, kc + 2 * BK, a0, b0, tid);
      asm volatile("s_wait_asynccnt 0x4" ::: "memory");
    } else {
      asm volatile("s_wait_asynccnt 0x0" ::: "memory");
    }
    __syncthreads();
    compute_tile(lsA1, lsB1, wm, wn, half, mr, acc);
    __syncthreads();
  }
}

// ---------------------------------------------------------------------------
// GEMM 1: H = tanh(Y @ W1^T-layout + b1 + t * w1row), output bf16
// ---------------------------------------------------------------------------
__global__ __launch_bounds__(256) void gemm_tanh_kernel(
    const __bf16* __restrict__ A, const __bf16* __restrict__ Bt,
    const float* __restrict__ bias, const float* __restrict__ wrow, float t,
    __bf16* __restrict__ Out, int N, int K) {
  __shared__ __attribute__((aligned(16))) __bf16 ls[4 * BM * LDSS];
  const int m0 = blockIdx.y * BM, n0 = blockIdx.x * BN;
  v8f acc[2][4] = {};
  gemm_mainloop(A, Bt, K, m0, n0, ls, acc);

  const int tid = threadIdx.x, lane = tid & 31, wave = tid >> 5;
  const int wm = wave & 3, wn = wave >> 2, half = lane >> 4;
#pragma unroll
  for (int j = 0; j < 4; ++j) {
    const int col = n0 + wn * 64 + j * 16 + (lane & 15);
    const float bb = bias[col] + t * wrow[col];
#pragma unroll
    for (int i = 0; i < 2; ++i) {
      const int rbase = m0 + wm * 32 + i * 16 + half * 8;
#pragma unroll
      for (int r = 0; r < 8; ++r) {
        const float v = tanhf(acc[i][j][r] + bb);
        Out[(size_t)(rbase + r) * N + col] = (__bf16)v;
      }
    }
  }
}

// ---------------------------------------------------------------------------
// GEMM 2 + fused RK update: k = H @ W2^T-layout + b2
//   yacc += coefA * k;   if STAGE: stage = bf16(yn + coefB * k)
// ---------------------------------------------------------------------------
template <bool STAGE>
__global__ __launch_bounds__(256) void gemm_k_kernel(
    const __bf16* __restrict__ A, const __bf16* __restrict__ Bt,
    const float* __restrict__ bias, const float* __restrict__ yn,
    float* __restrict__ yacc, float coefA, float coefB,
    __bf16* __restrict__ stage, int N, int K) {
  __shared__ __attribute__((aligned(16))) __bf16 ls[4 * BM * LDSS];
  const int m0 = blockIdx.y * BM, n0 = blockIdx.x * BN;
  v8f acc[2][4] = {};
  gemm_mainloop(A, Bt, K, m0, n0, ls, acc);

  const int tid = threadIdx.x, lane = tid & 31, wave = tid >> 5;
  const int wm = wave & 3, wn = wave >> 2, half = lane >> 4;
#pragma unroll
  for (int j = 0; j < 4; ++j) {
    const int col = n0 + wn * 64 + j * 16 + (lane & 15);
    const float bb = bias[col];
#pragma unroll
    for (int i = 0; i < 2; ++i) {
      const int rbase = m0 + wm * 32 + i * 16 + half * 8;
#pragma unroll
      for (int r = 0; r < 8; ++r) {
        const size_t idx = (size_t)(rbase + r) * N + col;
        const float kv = acc[i][j][r] + bb;
        yacc[idx] += coefA * kv;
        if (STAGE) stage[idx] = (__bf16)(yn[idx] + coefB * kv);
      }
    }
  }
}

// ---------------------------------------------------------------------------
// W (KxN fp32, row-major) -> WT (NxK bf16, row-major), tiled via LDS
// ---------------------------------------------------------------------------
__global__ __launch_bounds__(256) void transpose_convert_kernel(
    const float* __restrict__ in, __bf16* __restrict__ out, int K, int N) {
  __shared__ float tile[32][33];
  const int n0 = blockIdx.x * 32, k0 = blockIdx.y * 32;
  const int tx = threadIdx.x & 31, ty = threadIdx.x >> 5;
#pragma unroll
  for (int r = ty; r < 32; r += 8)
    tile[r][tx] = in[(size_t)(k0 + r) * N + n0 + tx];
  __syncthreads();
#pragma unroll
  for (int r = ty; r < 32; r += 8)
    out[(size_t)(n0 + r) * K + k0 + tx] = (__bf16)tile[tx][r];
}

// ---------------------------------------------------------------------------
// outf = in (fp32 copy), outb = bf16(in)   (vectorized x4)
// ---------------------------------------------------------------------------
__global__ __launch_bounds__(256) void copy_convert_kernel(
    const float* __restrict__ in, float* __restrict__ outf,
    __bf16* __restrict__ outb, size_t n) {
  const size_t i = ((size_t)blockIdx.x * 256 + threadIdx.x) * 4;
  if (i + 3 < n) {
    const float4 v = *(const float4*)&in[i];
    *(float4*)&outf[i] = v;
    v4bf b;
    b[0] = (__bf16)v.x; b[1] = (__bf16)v.y;
    b[2] = (__bf16)v.z; b[3] = (__bf16)v.w;
    *(v4bf*)&outb[i] = b;
  }
}

extern "C" void kernel_launch(void* const* d_in, const int* in_sizes, int n_in,
                              void* d_out, int out_size, void* d_ws,
                              size_t ws_size, hipStream_t stream) {
  const float* x  = (const float*)d_in[0];  // [4096, 2048]
  const float* W1 = (const float*)d_in[1];  // [2049, 2048]
  const float* b1 = (const float*)d_in[2];  // [2048]
  const float* W2 = (const float*)d_in[3];  // [2048, 2048]
  const float* b2 = (const float*)d_in[4];  // [2048]
  float* out = (float*)d_out;

  const int Bm = 4096, D = 2048, H = 2048;
  const float* w1row = W1 + (size_t)D * H;  // time row of W1

  char* w = (char*)d_ws;
  __bf16* W1T = (__bf16*)w; w += (size_t)D * H * 2;   // [H][D]
  __bf16* W2T = (__bf16*)w; w += (size_t)H * D * 2;   // [D][H]
  __bf16* stg = (__bf16*)w; w += (size_t)Bm * D * 2;  // stage input (bf16)
  __bf16* Hm  = (__bf16*)w; w += (size_t)Bm * H * 2;  // hidden (bf16)
  float*  y0  = (float*)w;  w += (size_t)Bm * D * 4;  // state after step 1

  const dim3 blk(256);
  const dim3 tg(H / 32, D / 32);              // both weights are 2048x2048
  const dim3 gg(D / BN, Bm / BM);             // GEMM grid (N/128, M/128)
  const size_t nel = (size_t)Bm * D;
  const dim3 cg((unsigned)(nel / 4 / 256));
  const float h = 0.5f;

  // Weights: fp32 [K][N] -> bf16 [N][K] once per launch
  transpose_convert_kernel<<<tg, blk, 0, stream>>>(W1, W1T, D, H);
  transpose_convert_kernel<<<tg, blk, 0, stream>>>(W2, W2T, H, D);

  // ---- RK4 step 1: t0 = 0, yn = x, yacc = y0 ----
  copy_convert_kernel<<<cg, blk, 0, stream>>>(x, y0, stg, nel);
  // k1
  gemm_tanh_kernel<<<gg, blk, 0, stream>>>(stg, W1T, b1, w1row, 0.00f, Hm, H, D);
  gemm_k_kernel<true><<<gg, blk, 0, stream>>>(Hm, W2T, b2, x, y0, h / 6, h / 2, stg, D, H);
  // k2
  gemm_tanh_kernel<<<gg, blk, 0, stream>>>(stg, W1T, b1, w1row, 0.25f, Hm, H, D);
  gemm_k_kernel<true><<<gg, blk, 0, stream>>>(Hm, W2T, b2, x, y0, h / 3, h / 2, stg, D, H);
  // k3
  gemm_tanh_kernel<<<gg, blk, 0, stream>>>(stg, W1T, b1, w1row, 0.25f, Hm, H, D);
  gemm_k_kernel<true><<<gg, blk, 0, stream>>>(Hm, W2T, b2, x, y0, h / 3, h, stg, D, H);
  // k4
  gemm_tanh_kernel<<<gg, blk, 0, stream>>>(stg, W1T, b1, w1row, 0.50f, Hm, H, D);
  gemm_k_kernel<false><<<gg, blk, 0, stream>>>(Hm, W2T, b2, y0, y0, h / 6, 0.f, nullptr, D, H);

  // ---- RK4 step 2: t0 = 0.5, yn = y0, yacc = out ----
  copy_convert_kernel<<<cg, blk, 0, stream>>>(y0, out, stg, nel);
  // k1
  gemm_tanh_kernel<<<gg, blk, 0, stream>>>(stg, W1T, b1, w1row, 0.50f, Hm, H, D);
  gemm_k_kernel<true><<<gg, blk, 0, stream>>>(Hm, W2T, b2, y0, out, h / 6, h / 2, stg, D, H);
  // k2
  gemm_tanh_kernel<<<gg, blk, 0, stream>>>(stg, W1T, b1, w1row, 0.75f, Hm, H, D);
  gemm_k_kernel<true><<<gg, blk, 0, stream>>>(Hm, W2T, b2, y0, out, h / 3, h / 2, stg, D, H);
  // k3
  gemm_tanh_kernel<<<gg, blk, 0, stream>>>(stg, W1T, b1, w1row, 0.75f, Hm, H, D);
  gemm_k_kernel<true><<<gg, blk, 0, stream>>>(Hm, W2T, b2, y0, out, h / 3, h, stg, D, H);
  // k4
  gemm_tanh_kernel<<<gg, blk, 0, stream>>>(stg, W1T, b1, w1row, 1.00f, Hm, H, D);
  gemm_k_kernel<false><<<gg, blk, 0, stream>>>(Hm, W2T, b2, out, out, h / 6, 0.f, nullptr, D, H);
}
